// PointCloudResNet_42399917146317
// MI455X (gfx1250) — matrix operand order repaired and checked by reading, not compiled, not run on hardware
//
#include <hip/hip_runtime.h>
#include <hip/hip_bf16.h>

typedef _Float16 half_t;
typedef __attribute__((ext_vector_type(16))) _Float16 v16h;
typedef __attribute__((ext_vector_type(8)))  float    v8f;

#define RDIM  16
#define GVOX  4096
#define KNNK  8
#define HDIM  64
#define BATCH 4
#define NPTS  4096
#define WTS   72   // padded LDS row stride (halves) for weight tiles

#if defined(__HIP_DEVICE_COMPILE__) && __has_builtin(__builtin_amdgcn_tensor_load_to_lds)
#define HAVE_TDM 1
#else
#define HAVE_TDM 0
#endif

__device__ __forceinline__ float gelu_f(float x) {
    float x3 = x * x * x;
    return 0.5f * x * (1.0f + tanhf(0.79788456080286535588f * (x + 0.044715f * x3)));
}

// ---- WMMA fragment loaders from LDS (CDNA5 16x16x32 f16 layouts, ISA 7.12.2) ----
// A (16x32 f16): lane m (<16): K 0..7 (v0-3) and 16..23 (v4-7); lane m+16: K 8..15 / 24..31.
__device__ __forceinline__ v16h lds_load_afrag(const half_t* base, int stride, int kb) {
    int lane = threadIdx.x & 31;
    int r = lane & 15, hi = lane >> 4;
    const half_t* p0 = base + r * stride + kb + hi * 8;
    v16h a;
#pragma unroll
    for (int i = 0; i < 8; i++) { a[i] = p0[i]; a[i + 8] = p0[16 + i]; }
    return a;
}
// B (32x16 f16): lane n holds column n; lanes 0-15: K 0..7 / 16..23, lanes 16-31: K 8..15 / 24..31.
// Weights staged transposed: wt[n][k], row stride `stride` halves.
__device__ __forceinline__ v16h lds_load_bfrag(const half_t* wt, int stride, int ntl, int kb) {
    int lane = threadIdx.x & 31;
    int n = lane & 15, hi = lane >> 4;
    const half_t* p = wt + (ntl * 16 + n) * stride + kb + hi * 8;
    v16h b;
#pragma unroll
    for (int i = 0; i < 8; i++) { b[i] = p[i]; b[i + 8] = p[16 + i]; }
    return b;
}

#if HAVE_TDM
typedef unsigned int v4u __attribute__((ext_vector_type(4)));
typedef int          v4i __attribute__((ext_vector_type(4)));
typedef int          v8i __attribute__((ext_vector_type(8)));

// TDM: DMA `nelem` f16 elements global->LDS as a 1-D tile, inserting a 4-dword pad
// after every 32 dwords (=> 64-half rows land at 72-half LDS stride).
// Descriptor packing per CDNA5 ISA.
__device__ __forceinline__ void tdm_load_f16_padded(unsigned lds_byte_off, const void* gsrc,
                                                    unsigned nelem) {
    unsigned long long ga = (unsigned long long)(size_t)gsrc;
    v4u g0;
    g0[0] = 1u;                                        // count=1, user-mode, no gather
    g0[1] = lds_byte_off;                              // lds_addr
    g0[2] = (unsigned)(ga & 0xffffffffu);              // global_addr[31:0]
    g0[3] = (unsigned)((ga >> 32) & 0x01ffffffu) | (2u << 30);  // addr[56:32] | type=2
    unsigned td0 = nelem;                              // tensor_dim0 (elements)
    unsigned td1 = 1u;                                 // tensor_dim1
    unsigned tile0 = nelem;                            // tile_dim0 (16-bit)
    v8i g1;
    g1[0] = (int)((1u << 16)        // data_size = 1 -> 2 bytes
                | (1u << 20)        // pad_enable
                | (4u << 22)        // pad_interval: 32 dwords
                | (3u << 25));      // pad_amount:   4 dwords
    g1[1] = (int)((td0 & 0xffffu) << 16);
    g1[2] = (int)((td0 >> 16) | ((td1 & 0xffffu) << 16));
    g1[3] = (int)((td1 >> 16) | ((tile0 & 0xffffu) << 16));
    g1[4] = 0;                                         // tile_dim1=0, tile_dim2=0 (1-D)
    g1[5] = (int)td0;                                  // tensor_dim0_stride lo
    g1[6] = 0;                                         // stride0 hi | stride1 lo
    g1[7] = 0;                                         // stride1 hi
    v4i z4 = {0, 0, 0, 0};
#if __clang_major__ >= 23
    v8i z8 = {0, 0, 0, 0, 0, 0, 0, 0};
    __builtin_amdgcn_tensor_load_to_lds(g0, g1, z4, z4, z8, 0);
#else
    __builtin_amdgcn_tensor_load_to_lds(g0, g1, z4, z4, 0);
#endif
}

__device__ __forceinline__ unsigned lds_byte_offset(const void* p) {
    // LDS aperture: flat address bits [31:0] carry the LDS byte offset.
    return (unsigned)(size_t)p;
}
#endif  // HAVE_TDM

// ============================ weight prep (f32 -> f16, transposed) ============================
// wbuf layout (halves): [0,4096) pe_w2^T [n][k]; [4096,8192) up_w^T [n][k];
// [8192, ...) conv weights as [layer(4)][tap(27)][o(64)][i(64)]
__global__ void prep_weights_kernel(const float* __restrict__ pe_w2, const float* __restrict__ up_w,
                                    const float* __restrict__ conv_w, half_t* __restrict__ wbuf) {
    int i = blockIdx.x * blockDim.x + threadIdx.x;
    if (i < 4096) {
        int n = i >> 6, k = i & 63;
        wbuf[i] = (half_t)pe_w2[k * 64 + n];
    } else if (i < 8192) {
        int j = i - 4096; int n = j >> 6, k = j & 63;
        wbuf[i] = (half_t)up_w[k * 64 + n];
    } else if (i < 8192 + 4 * 27 * 64 * 64) {
        int j = i - 8192;
        int ic  = j & 63;
        int o   = (j >> 6) & 63;
        int tap = (j >> 12) % 27;
        int lc  = j / (27 * 4096);
        wbuf[i] = (half_t)conv_w[(((size_t)lc * 64 + o) * 64 + ic) * 27 + tap];
    }
}

__global__ void zero_kernel(float* p, int n) {
    int i = blockIdx.x * blockDim.x + threadIdx.x;
    if (i < n) p[i] = 0.0f;
}

// ============================ brute-force KNN: one wave per voxel ============================
__global__ void knn_kernel(const float* __restrict__ pos, int* __restrict__ idx_out) {
    int wave = threadIdx.x >> 5;
    int lane = threadIdx.x & 31;
    int vox = blockIdx.x * 8 + wave;
    int b = vox >> 12;
    int g = vox & 4095;
    int gi = g >> 8, gj = (g >> 4) & 15, gk = g & 15;
    float gx = -1.0f + (2.0f / 15.0f) * gi;
    float gy = -1.0f + (2.0f / 15.0f) * gj;
    float gz = -1.0f + (2.0f / 15.0f) * gk;
    float bd[KNNK]; int bi[KNNK];
#pragma unroll
    for (int i = 0; i < KNNK; i++) { bd[i] = 3.0e38f; bi[i] = 0x7fffffff; }
    const float* pb = pos + (size_t)b * NPTS * 3;
    for (int n = lane; n < NPTS; n += 32) {
        float dx = pb[n * 3 + 0] - gx;
        float dy = pb[n * 3 + 1] - gy;
        float dz = pb[n * 3 + 2] - gz;
        float d = dx * dx + dy * dy + dz * dz;
        if (d < bd[KNNK - 1]) {
            int p = KNNK - 1;
            while (p > 0 && bd[p - 1] > d) { bd[p] = bd[p - 1]; bi[p] = bi[p - 1]; p--; }
            bd[p] = d; bi[p] = n;
        }
    }
    // cross-lane merge: 8 rounds of wave-wide argmin (tie-break on smaller point index)
    int ptr = 0;
    for (int k = 0; k < KNNK; k++) {
        float d = (ptr < KNNK) ? bd[ptr] : 3.0e38f;
        int   id = (ptr < KNNK) ? bi[ptr] : 0x7fffffff;
        int   ow = lane;
#pragma unroll
        for (int off = 16; off > 0; off >>= 1) {
            float d2 = __shfl_xor(d, off, 32);
            int   i2 = __shfl_xor(id, off, 32);
            int   o2 = __shfl_xor(ow, off, 32);
            if (d2 < d || (d2 == d && i2 < id)) { d = d2; id = i2; ow = o2; }
        }
        if (ow == lane) ptr++;
        if (lane == 0) idx_out[(size_t)vox * KNNK + k] = id;
    }
}

// ============== gather + pos-embed MLP layer1 + WMMA (t1 @ pe_w2) + msg * mean ==============
// Block: 256 thr / 8 waves; handles 8 voxels = 64 rows (voxel-major, 8 neighbors each).
// Weight tile is TDM-DMA'd into LDS by wave 0, overlapped with the phase-1 VALU work.
__global__ void msg_mean_kernel(const float* __restrict__ pos, const float* __restrict__ xin,
                                const int* __restrict__ idx,
                                const float* __restrict__ pe_w1, const float* __restrict__ pe_b1,
                                const float* __restrict__ pe_b2,
                                const float* __restrict__ f_w, const float* __restrict__ f_b,
                                const half_t* __restrict__ w2t, float* __restrict__ agg) {
    __shared__ half_t T1[64 * 72];
    __shared__ float  Fm[64 * 68];
    __shared__ half_t Wt[64 * WTS];
    int tid = threadIdx.x;
    int wv = tid >> 5, lane = tid & 31;
    int vox0 = blockIdx.x * 8;
#if HAVE_TDM
    if (wv == 0) tdm_load_f16_padded(lds_byte_offset(&Wt[0]), w2t, 4096);
#else
    { // stage transposed pe_w2 manually
        int base = tid * 16; int r = base >> 6, c = base & 63;
#pragma unroll
        for (int q = 0; q < 16; q++) Wt[r * WTS + c + q] = w2t[base + q];
    }
#endif
    { // phase 1: t1 = gelu(rel @ pe_w1 + b1); fm = feat @ f_w + f_b
        int row = tid >> 2;             // 0..63
        int c0 = (tid & 3) * 16;
        int vloc = row >> 3, kk = row & 7;
        int vox = vox0 + vloc;
        int b = vox >> 12, g = vox & 4095;
        int gi = g >> 8, gj = (g >> 4) & 15, gk = g & 15;
        float gx = -1.0f + (2.0f / 15.0f) * gi;
        float gy = -1.0f + (2.0f / 15.0f) * gj;
        float gz = -1.0f + (2.0f / 15.0f) * gk;
        int nb = idx[(size_t)vox * KNNK + kk];
        const float* pp = pos + ((size_t)b * NPTS + nb) * 3;
        const float* ff = xin + ((size_t)b * NPTS + nb) * 3;
        float r0 = pp[0] - gx, r1 = pp[1] - gy, r2 = pp[2] - gz;
        float f0 = ff[0], f1 = ff[1], f2 = ff[2];
#pragma unroll
        for (int q = 0; q < 16; q++) {
            int c = c0 + q;
            float t = r0 * pe_w1[c] + r1 * pe_w1[64 + c] + r2 * pe_w1[128 + c] + pe_b1[c];
            T1[row * 72 + c] = (half_t)gelu_f(t);
            Fm[row * 68 + c] = f0 * f_w[c] + f1 * f_w[64 + c] + f2 * f_w[128 + c] + f_b[c];
        }
    }
#if HAVE_TDM
    if (wv == 0) __builtin_amdgcn_s_wait_tensorcnt(0);
#endif
    __syncthreads();
    int np = lane & 15, hi = lane >> 4;
#pragma unroll
    for (int jj = 0; jj < 2; jj++) {
        int job = wv * 2 + jj;
        int mt = job >> 2, ntl = job & 3;
        int col = ntl * 16 + np;
        v8f c;
        float bias = pe_b2[col];
#pragma unroll
        for (int q = 0; q < 8; q++) c[q] = bias;
#pragma unroll
        for (int kb = 0; kb < 64; kb += 32) {
            v16h a = lds_load_afrag(T1 + mt * 16 * 72, 72, kb);
            v16h bb = lds_load_bfrag(Wt, WTS, ntl, kb);
            c = __builtin_amdgcn_wmma_f32_16x16x32_f16(false, a, false, bb, (short)0, c, false, false);
        }
        // msg = pe * fm; mean over 8 neighbor rows (VGPR dimension)
        float s = 0.0f;
        int rbase = mt * 16 + hi * 8;
#pragma unroll
        for (int q = 0; q < 8; q++) s += c[q] * Fm[(rbase + q) * 68 + col];
        int vox = vox0 + mt * 2 + hi;
        agg[(size_t)vox * 64 + col] = s * (1.0f / KNNK);
    }
}

// ============== update net: h = gelu(agg @ up_w + b); scatter to NCDHW-as-channels-last ==============
__global__ void up_kernel(const float* __restrict__ agg, const half_t* __restrict__ upwt,
                          const float* __restrict__ up_b, float* __restrict__ vol) {
    __shared__ half_t A[64 * 72];
    __shared__ half_t Wt[64 * WTS];
    int tid = threadIdx.x;
    int wv = tid >> 5, lane = tid & 31;
    int m0 = blockIdx.x * 64;
#if HAVE_TDM
    if (wv == 0) tdm_load_f16_padded(lds_byte_offset(&Wt[0]), upwt, 4096);
#else
    {
        int base = tid * 16; int r = base >> 6, c = base & 63;
#pragma unroll
        for (int q = 0; q < 16; q++) Wt[r * WTS + c + q] = upwt[base + q];
    }
#endif
    {
        int row = tid >> 2, c0 = (tid & 3) * 16;
        const float* src = agg + ((size_t)(m0 + row)) * 64 + c0;
#pragma unroll
        for (int q = 0; q < 16; q++) A[row * 72 + c0 + q] = (half_t)src[q];
    }
#if HAVE_TDM
    if (wv == 0) __builtin_amdgcn_s_wait_tensorcnt(0);
#endif
    __syncthreads();
    int np = lane & 15, hi = lane >> 4;
#pragma unroll
    for (int jj = 0; jj < 2; jj++) {
        int job = wv * 2 + jj, mt = job >> 2, ntl = job & 3;
        int col = ntl * 16 + np;
        v8f c; float bias = up_b[col];
#pragma unroll
        for (int q = 0; q < 8; q++) c[q] = bias;
#pragma unroll
        for (int kb = 0; kb < 64; kb += 32) {
            v16h a = lds_load_afrag(A + mt * 16 * 72, 72, kb);
            v16h bb = lds_load_bfrag(Wt, WTS, ntl, kb);
            c = __builtin_amdgcn_wmma_f32_16x16x32_f16(false, a, false, bb, (short)0, c, false, false);
        }
#pragma unroll
        for (int q = 0; q < 8; q++) {
            int m = m0 + mt * 16 + hi * 8 + q;     // m = b*G + g,  g = gi*256 + gj*16 + gk
            int b = m >> 12, g = m & 4095;
            int gi = g >> 8, gj = (g >> 4) & 15, gk = g & 15;
            int s = (gk * 16 + gj) * 16 + gi;      // D=gk, H=gj, W=gi (the transpose(0,4,3,2,1))
            vol[((size_t)(b * GVOX + s)) * 64 + col] = gelu_f(c[q]);
        }
    }
}

// ====== 3x3x3 conv as 27 shifted 64x64 WMMA GEMMs; TDM double-buffered weight streaming ======
// Block: 256 thr / 8 waves; M-tile = 32 rows (2 mtiles x 4 ntiles = 8 wave jobs).
__global__ void conv_kernel(const float* __restrict__ in, const half_t* __restrict__ wh,
                            const float* __restrict__ bias, float* __restrict__ out,
                            float* __restrict__ stats, int D, int HH, int WW) {
    __shared__ half_t A[32 * 72];
#if HAVE_TDM
    __shared__ half_t Wt[2][64 * WTS];
#else
    __shared__ half_t Wt[1][64 * WTS];
#endif
    int tid = threadIdx.x;
    int S = D * HH * WW;
    int m0 = blockIdx.x * 32;
    int wv = tid >> 5, lane = tid & 31, np = lane & 15, hi = lane >> 4;
    int mt = wv >> 2, ntl = wv & 3;
    int col = ntl * 16 + np;
    v8f c;
#pragma unroll
    for (int q = 0; q < 8; q++) c[q] = 0.0f;
    // staging-thread row decode (8 threads per row, 8 channels each)
    int ar = tid >> 3;
    int ac0 = (tid & 7) * 8;
    int m = m0 + ar;
    int b = m / S; int s = m % S;
    int z = s / (HH * WW); int yy = (s / WW) % HH; int xx = s % WW;

#if HAVE_TDM
    if (wv == 0) tdm_load_f16_padded(lds_byte_offset(&Wt[0][0]), wh, 4096);
#endif
    for (int tap = 0; tap < 27; tap++) {
        int dz = tap / 9 - 1, dy = (tap / 3) % 3 - 1, dx = tap % 3 - 1;
        int zs = z + dz, ys = yy + dy, xs = xx + dx;
        bool ok = (zs >= 0) && (zs < D) && (ys >= 0) && (ys < HH) && (xs >= 0) && (xs < WW);
        const float* src = in + ((size_t)(b * S + (zs * HH + ys) * WW + xs)) * 64 + ac0;
#pragma unroll
        for (int q = 0; q < 8; q++) A[ar * 72 + ac0 + q] = ok ? (half_t)src[q] : (half_t)0.0f;
#if HAVE_TDM
        if (wv == 0) __builtin_amdgcn_s_wait_tensorcnt(0);  // current buffer landed
        __syncthreads();                                     // A + Wt[tap&1] published
        if (wv == 0 && tap < 26)                             // prefetch next tap's weights
            tdm_load_f16_padded(lds_byte_offset(&Wt[(tap + 1) & 1][0]),
                                wh + (size_t)(tap + 1) * 4096, 4096);
        const half_t* wt = &Wt[tap & 1][0];
#else
        {
            const half_t* wtap = wh + tap * 4096;
            int base = tid * 16; int r = base >> 6, cc = base & 63;
#pragma unroll
            for (int q = 0; q < 16; q++) Wt[0][r * WTS + cc + q] = wtap[base + q];
        }
        __syncthreads();
        const half_t* wt = &Wt[0][0];
#endif
#pragma unroll
        for (int kb = 0; kb < 64; kb += 32) {
            v16h a = lds_load_afrag(A + mt * 16 * 72, 72, kb);
            v16h bb = lds_load_bfrag(wt, WTS, ntl, kb);
            c = __builtin_amdgcn_wmma_f32_16x16x32_f16(false, a, false, bb, (short)0, c, false, false);
        }
        __syncthreads();
    }
    float bsum = 0.0f, bss = 0.0f;
    float bcol = bias[col];
#pragma unroll
    for (int q = 0; q < 8; q++) {
        float v = c[q] + bcol;
        int mr = m0 + mt * 16 + hi * 8 + q;
        out[(size_t)mr * 64 + col] = v;
        bsum += v; bss += v * v;
    }
    atomicAdd(&stats[col], bsum);
    atomicAdd(&stats[64 + col], bss);
}

__global__ void bn_relu_kernel(const float* __restrict__ y, const float* __restrict__ stats,
                               float cnt_inv, float* __restrict__ out, int total) {
    int i = blockIdx.x * blockDim.x + threadIdx.x;
    if (i >= total) return;
    int ch = i & 63;
    float m = stats[ch] * cnt_inv;
    float v = stats[64 + ch] * cnt_inv - m * m;
    float r = rsqrtf(v + 1e-5f);
    out[i] = fmaxf((y[i] - m) * r, 0.0f);
}

__global__ void bn_residual_kernel(const float* __restrict__ y, const float* __restrict__ stats,
                                   float cnt_inv, const float* __restrict__ resid,
                                   float* __restrict__ out, int total, float* __restrict__ fstats) {
    int i = blockIdx.x * blockDim.x + threadIdx.x;
    if (i >= total) return;
    int ch = i & 63;
    float m = stats[ch] * cnt_inv;
    float v = stats[64 + ch] * cnt_inv - m * m;
    float r = rsqrtf(v + 1e-5f);
    float o = fmaxf(resid[i] + (y[i] - m) * r, 0.0f);
    out[i] = o;
    if (fstats) { atomicAdd(&fstats[ch], o); atomicAdd(&fstats[64 + ch], o * o); }
}

__global__ void maxpool_kernel(const float* __restrict__ in, float* __restrict__ out) {
    int i = blockIdx.x * blockDim.x + threadIdx.x;
    if (i >= BATCH * 512 * 64) return;
    int ch = i & 63; int s = (i >> 6) & 511; int b = i >> 15;
    int zo = s >> 6, yo = (s >> 3) & 7, xo = s & 7;
    float mx = -3.0e38f;
#pragma unroll
    for (int d2 = 0; d2 < 2; d2++)
#pragma unroll
        for (int y2 = 0; y2 < 2; y2++)
#pragma unroll
            for (int x2 = 0; x2 < 2; x2++) {
                int z = zo * 2 + d2, yv = yo * 2 + y2, xv = xo * 2 + x2;
                mx = fmaxf(mx, in[((size_t)(b * GVOX + (z * 16 + yv) * 16 + xv)) * 64 + ch]);
            }
    out[i] = mx;
}

// final BN (affine) + mean pool (linearity lets BN commute with mean) + linear head
__global__ void readout_kernel(const float* __restrict__ vol2, const float* __restrict__ fstats,
                               const float* __restrict__ on_g, const float* __restrict__ on_b,
                               const float* __restrict__ ro_w, const float* __restrict__ ro_b,
                               float* __restrict__ outp) {
    __shared__ float pooled[BATCH * 64];
    int t = threadIdx.x;
    int b = t >> 6, ch = t & 63;
    float sum = 0.0f;
    for (int s = 0; s < 512; s++) sum += vol2[((size_t)(b * 512 + s)) * 64 + ch];
    float cnt_inv = 1.0f / 2048.0f;
    float m = fstats[ch] * cnt_inv;
    float v = fstats[64 + ch] * cnt_inv - m * m;
    float r = rsqrtf(v + 1e-5f);
    pooled[t] = ((sum * (1.0f / 512.0f)) - m) * r * on_g[ch] + on_b[ch];
    __syncthreads();
    if (t < 64) {
        int bb = t >> 4, o = t & 15;
        float acc = ro_b[o];
        for (int cc = 0; cc < 64; cc++) acc += pooled[bb * 64 + cc] * ro_w[cc * 16 + o];
        outp[bb * 16 + o] = acc;
    }
}

extern "C" void kernel_launch(void* const* d_in, const int* in_sizes, int n_in,
                              void* d_out, int out_size, void* d_ws, size_t ws_size,
                              hipStream_t stream) {
    (void)in_sizes; (void)n_in; (void)out_size; (void)ws_size;
    const float* pos    = (const float*)d_in[0];
    const float* x      = (const float*)d_in[1];
    const float* pe_w1  = (const float*)d_in[2];
    const float* pe_b1  = (const float*)d_in[3];
    const float* pe_w2  = (const float*)d_in[4];
    const float* pe_b2  = (const float*)d_in[5];
    const float* f_w    = (const float*)d_in[6];
    const float* f_b    = (const float*)d_in[7];
    const float* up_w   = (const float*)d_in[8];
    const float* up_b   = (const float*)d_in[9];
    const float* conv_w = (const float*)d_in[10];
    const float* conv_b = (const float*)d_in[11];
    const float* on_g   = (const float*)d_in[12];
    const float* on_b   = (const float*)d_in[13];
    const float* ro_w   = (const float*)d_in[14];
    const float* ro_b   = (const float*)d_in[15];

    char* ws = (char*)d_ws;
    int*    idx   = (int*)ws;                         // 4*4096*8*4   = 512 KB
    half_t* wbuf  = (half_t*)(ws + 524288);           // 450560 halves ~ 880 KB
    float*  stats = (float*)(ws + 1441792);           // 5 x 128 f32
    float*  bufA  = (float*)(ws + 1445888);           // 4 MB
    float*  bufB  = bufA + (1 << 20);                 // 4 MB
    float*  bufC  = bufB + (1 << 20);                 // 4 MB
    half_t* convh = wbuf + 8192;

    prep_weights_kernel<<<1760, 256, 0, stream>>>(pe_w2, up_w, conv_w, wbuf);
    zero_kernel<<<3, 256, 0, stream>>>(stats, 5 * 128);
    knn_kernel<<<2048, 256, 0, stream>>>(pos, idx);
    msg_mean_kernel<<<2048, 256, 0, stream>>>(pos, x, idx, pe_w1, pe_b1, pe_b2, f_w, f_b,
                                              wbuf, bufA);
    up_kernel<<<256, 256, 0, stream>>>(bufA, wbuf + 4096, up_b, bufB);
    // ---- block 0 (R=16, M=16384) ----
    conv_kernel<<<512, 256, 0, stream>>>(bufB, convh + 0 * 27 * 4096, conv_b + 0, bufA,
                                         stats + 0, 16, 16, 16);
    bn_relu_kernel<<<4096, 256, 0, stream>>>(bufA, stats + 0, 1.0f / 16384.0f, bufC, 16384 * 64);
    conv_kernel<<<512, 256, 0, stream>>>(bufC, convh + 1 * 27 * 4096, conv_b + 64, bufA,
                                         stats + 128, 16, 16, 16);
    bn_residual_kernel<<<4096, 256, 0, stream>>>(bufA, stats + 128, 1.0f / 16384.0f, bufB, bufC,
                                                 16384 * 64, nullptr);
    maxpool_kernel<<<512, 256, 0, stream>>>(bufC, bufB);
    // ---- block 1 (R=8, M=2048) ----
    conv_kernel<<<64, 256, 0, stream>>>(bufB, convh + 2 * 27 * 4096, conv_b + 128, bufA,
                                        stats + 256, 8, 8, 8);
    bn_relu_kernel<<<512, 256, 0, stream>>>(bufA, stats + 256, 1.0f / 2048.0f, bufC, 2048 * 64);
    conv_kernel<<<64, 256, 0, stream>>>(bufC, convh + 3 * 27 * 4096, conv_b + 192, bufA,
                                        stats + 384, 8, 8, 8);
    bn_residual_kernel<<<512, 256, 0, stream>>>(bufA, stats + 384, 1.0f / 2048.0f, bufB, bufC,
                                                2048 * 64, stats + 512);
    readout_kernel<<<1, 256, 0, stream>>>(bufC, stats + 512, on_g, on_b, ro_w, ro_b,
                                          (float*)d_out);
}
